// CGCNNFormation_7292854469256
// MI455X (gfx1250) — compile-verified, matrix operand order checked by reading, not run on hardware
//
#include <hip/hip_runtime.h>
#include <hip/hip_bf16.h>
#include <stdint.h>

typedef __attribute__((ext_vector_type(16))) _Float16 v16h;
typedef __attribute__((ext_vector_type(4)))  _Float16 v4h;
typedef __attribute__((ext_vector_type(8)))  float    v8f;

#define N_NODES 50000
#define N_EDGES 800000
#define N_GRAPHS 256
#define FEA 64
#define EDIM 41
#define ZDIM 169
#define N_CONV 4

static_assert(N_NODES % 16 == 0, "node tiles exact");
static_assert(N_EDGES % 16 == 0, "edge tiles exact");

// One 16x16x32 WMMA k-step fragment = 32 lanes x 16 f16 = 512 f16.
#define FRAG_ELEMS 512

__device__ __forceinline__ v8f wmma_f16(v16h a, v16h b, v8f c) {
    return __builtin_amdgcn_wmma_f32_16x16x32_f16(false, a, false, b,
                                                  (short)0, c, false, false);
}

__device__ __forceinline__ float softplus_f(float x) {
    return (x > 20.0f) ? x : log1pf(__expf(x));
}
__device__ __forceinline__ float sigmoid_f(float x) {
    return 1.0f / (1.0f + __expf(-x));
}

// Element (n, k) -> flat index in packed A-fragment array (CDNA5 7.12.2).
__device__ __forceinline__ long afrag_index(int n, int k, int nKs) {
    int tile = n >> 4;
    int ks   = k >> 5;
    int kk   = k & 31;
    int kh   = (kk & 8) ? 1 : 0;                 // lane half
    int i    = (kk & 7) + ((kk & 16) ? 8 : 0);   // element within lane
    int lane = kh * 16 + (n & 15);
    return ((long)(tile * nKs + ks) * 32 + lane) * 16 + i;
}

// ---------------------------------------------------------------------------
// Pack fp32 row-major [M,K] into f16 A-fragments (K zero-padded to 32*NKS).
// K/NKS are compile-time so the pad guards fold away.
// ---------------------------------------------------------------------------
template <int K, int NKS>
__global__ void pack_a_kernel(const float* __restrict__ src, _Float16* __restrict__ dst,
                              int M) {
    long t = (long)blockIdx.x * blockDim.x + threadIdx.x;
    long total = (long)(M >> 4) * NKS * 32;
    if (t >= total) return;
    int  lane = (int)(t & 31);
    long tk   = t >> 5;
    int  ks   = (int)(tk % NKS);
    long tile = tk / NKS;
    int  kh   = lane >> 4;
    int  row  = (int)(tile * 16) + (lane & 15);
    const float* Ar = src + (long)row * K;
    v16h a;
#pragma unroll
    for (int i = 0; i < 16; ++i) {
        int k = ks * 32 + kh * 8 + (i & 7) + ((i & 8) ? 16 : 0);
        a[i] = (k < K) ? (_Float16)Ar[k] : (_Float16)0.0f;
    }
    *(v16h*)(dst + t * 16) = a;
}

// ---------------------------------------------------------------------------
// Pack fp32 row-major weights [*, ldb] (rows rowoff..rowoff+KLIM) into f16
// B-fragments for NCT column tiles x NKS k-steps (zero-padded k).
// ---------------------------------------------------------------------------
template <int KLIM, int NKS, int NCT>
__global__ void pack_b_kernel(const float* __restrict__ src, _Float16* __restrict__ dst,
                              int ldb, int rowoff) {
    int t = blockIdx.x * blockDim.x + threadIdx.x;
    if (t >= NCT * NKS * 32) return;
    int lane = t & 31;
    int ks = (t >> 5) % NKS;
    int ct = (t >> 5) / NKS;
    int kh = lane >> 4;
    int col = ct * 16 + (lane & 15);
    v16h b;
#pragma unroll
    for (int i = 0; i < 16; ++i) {
        int k = ks * 32 + kh * 8 + (i & 7) + ((i & 8) ? 16 : 0);
        b[i] = (k < KLIM) ? (_Float16)src[(long)(rowoff + k) * ldb + col]
                          : (_Float16)0.0f;
    }
    *(v16h*)(dst + (long)t * 16) = b;
}

// ---------------------------------------------------------------------------
// h = x @ emb_w + emb_b from packed fragments; writes fp32 h and f16 hpack.
// 1 block = 16 rows, 4 waves = 4 column tiles. nKs(x)=3, nKs(h)=2.
// ---------------------------------------------------------------------------
__global__ void embed_kernel(const _Float16* __restrict__ xp,
                             const _Float16* __restrict__ wp,
                             const float* __restrict__ bias,
                             float* __restrict__ h, _Float16* __restrict__ hp) {
    const int ct = threadIdx.x >> 5;
    const int lane = threadIdx.x & 31;
    const long tile = blockIdx.x;
    v8f acc = {};
#pragma unroll
    for (int s = 0; s < 3; ++s) {
        v16h a = *(const v16h*)(xp + ((tile * 3 + s) * 32 + lane) * 16);
        v16h b = *(const v16h*)(wp + (((long)ct * 3 + s) * 32 + lane) * 16);
        acc = wmma_f16(a, b, acc);
    }
    const int col = ct * 16 + (lane & 15);
    const float bv = bias[col];
    const int half = lane >> 4;
#pragma unroll
    for (int r = 0; r < 8; ++r) {
        int rr = (int)tile * 16 + half * 8 + r;   // always < N_NODES (exact tiles)
        float v = acc[r] + bv;
        h[(long)rr * FEA + col] = v;
        hp[afrag_index(rr, col, 2)] = (_Float16)v;
    }
}

// ---------------------------------------------------------------------------
// P[M,256] = h @ [Wf_i|Wf_j|Ws_i|Ws_j] (+bf on Fi, +bs on Si), all packed.
// One wave per 16x16 tile; 16 column tiles per 16-row tile. Exact grid.
// ---------------------------------------------------------------------------
__global__ void node_gemm_kernel(const _Float16* __restrict__ hp,
                                 const _Float16* __restrict__ wp,   // 16ct x 2ks frags
                                 const float* __restrict__ bf,
                                 const float* __restrict__ bs,
                                 float* __restrict__ P) {
    const int wid = blockIdx.x * (blockDim.x >> 5) + (threadIdx.x >> 5);
    const int mt = wid >> 4;
    const int ct = wid & 15;
    const int lane = threadIdx.x & 31;
    v8f acc = {};
#pragma unroll
    for (int s = 0; s < 2; ++s) {
        v16h a = *(const v16h*)(hp + ((long)(mt * 2 + s) * 32 + lane) * 16);
        v16h b = *(const v16h*)(wp + ((long)(ct * 2 + s) * 32 + lane) * 16);
        acc = wmma_f16(a, b, acc);
    }
    const int part = ct >> 2;                 // 0:Fi 1:Fj 2:Si 3:Sj
    const int col = (ct & 3) * 16 + (lane & 15);
    const float bv = (part == 0) ? bf[col] : ((part == 2) ? bs[col] : 0.0f);
    const int half = lane >> 4;
    const int ocol = part * 64 + col;
#pragma unroll
    for (int r = 0; r < 8; ++r) {
        int rr = mt * 16 + half * 8 + r;
        P[(long)rr * 256 + ocol] = acc[r] + bv;
    }
}

// ---------------------------------------------------------------------------
// Per-edge: one wave owns 16 edges x all 64 features (16 WMMA: gate+core x
// 4 col tiles x 2 k-steps, A-fragments loaded ONCE). Indices loaded as int4
// pairs; P gathers are L2-resident; f32 atomic scatter into agg.
// ---------------------------------------------------------------------------
__global__ void edge_kernel(const _Float16* __restrict__ ep,   // A frags, nKs=2
                            const int* __restrict__ eidx,      // [2,E]
                            const _Float16* __restrict__ wgp,  // gate B frags 4ct x 2ks
                            const _Float16* __restrict__ wsp,  // core B frags 4ct x 2ks
                            const float* __restrict__ P,       // [N,256]
                            float* __restrict__ agg) {         // [N,64]
    const int et = blockIdx.x * (blockDim.x >> 5) + (threadIdx.x >> 5);
    const int lane = threadIdx.x & 31;

    // prefetch next edge tile's fragment block (global_prefetch_b8)
    __builtin_prefetch(ep + ((long)(et + 1) * 2 * 32 + lane) * 16, 0, 0);

    v8f cg[4] = {{}, {}, {}, {}};
    v8f cs[4] = {{}, {}, {}, {}};
#pragma unroll
    for (int s = 0; s < 2; ++s) {
        v16h a = *(const v16h*)(ep + ((long)(et * 2 + s) * 32 + lane) * 16);
#pragma unroll
        for (int ct = 0; ct < 4; ++ct) {
            v16h bg = *(const v16h*)(wgp + ((ct * 2 + s) * 32 + lane) * 16);
            cg[ct] = wmma_f16(a, bg, cg[ct]);
            v16h bc = *(const v16h*)(wsp + ((ct * 2 + s) * 32 + lane) * 16);
            cs[ct] = wmma_f16(a, bc, cs[ct]);
        }
    }

    const int half = lane >> 4;
    const int eBase = et * 16 + half * 8;     // 8 consecutive edges for this lane
    int sarr[8], darr[8];
    {
        int4 s0 = *(const int4*)(eidx + eBase);
        int4 s1 = *(const int4*)(eidx + eBase + 4);
        int4 d0 = *(const int4*)(eidx + N_EDGES + eBase);
        int4 d1 = *(const int4*)(eidx + N_EDGES + eBase + 4);
        sarr[0] = s0.x; sarr[1] = s0.y; sarr[2] = s0.z; sarr[3] = s0.w;
        sarr[4] = s1.x; sarr[5] = s1.y; sarr[6] = s1.z; sarr[7] = s1.w;
        darr[0] = d0.x; darr[1] = d0.y; darr[2] = d0.z; darr[3] = d0.w;
        darr[4] = d1.x; darr[5] = d1.y; darr[6] = d1.z; darr[7] = d1.w;
    }

    const int cbase = lane & 15;
#pragma unroll
    for (int r = 0; r < 8; ++r) {
        const int d = darr[r];
        const int s = sarr[r];
        const float* Pd = P + (long)d * 256 + cbase;
        const float* Ps = P + (long)s * 256 + cbase;
        float* aggd = agg + (long)d * FEA + cbase;
#pragma unroll
        for (int ct = 0; ct < 4; ++ct) {
            const int c = ct * 16;
            float gpre = cg[ct][r] + Pd[c]       + Ps[64 + c];
            float spre = cs[ct][r] + Pd[128 + c] + Ps[192 + c];
            atomicAdd(aggd + c, sigmoid_f(gpre) * softplus_f(spre));
        }
    }
}

// ---------------------------------------------------------------------------
// h += BN(agg); refresh f16 hpack; re-zero agg. 4 elements per thread
// (a 4-aligned column group is contiguous in the A-fragment layout).
// ---------------------------------------------------------------------------
__global__ void bn_residual_kernel(float* __restrict__ h, _Float16* __restrict__ hp,
                                   float* __restrict__ agg,
                                   const float* __restrict__ g,
                                   const float* __restrict__ be,
                                   const float* __restrict__ mu,
                                   const float* __restrict__ var) {
    long q = (long)blockIdx.x * blockDim.x + threadIdx.x;   // N_NODES*16 quads
    long i = q * 4;
    int c = (int)(i & 63);
    int n = (int)(i >> 6);
    float4 gg = *(const float4*)(g + c);
    float4 bb = *(const float4*)(be + c);
    float4 mm = *(const float4*)(mu + c);
    float4 vv = *(const float4*)(var + c);
    float4 a  = *(const float4*)(agg + i);
    float4 hv = *(const float4*)(h + i);
    float4 o;
    o.x = hv.x + gg.x * (a.x - mm.x) * rsqrtf(vv.x + 1e-5f) + bb.x;
    o.y = hv.y + gg.y * (a.y - mm.y) * rsqrtf(vv.y + 1e-5f) + bb.y;
    o.z = hv.z + gg.z * (a.z - mm.z) * rsqrtf(vv.z + 1e-5f) + bb.z;
    o.w = hv.w + gg.w * (a.w - mm.w) * rsqrtf(vv.w + 1e-5f) + bb.w;
    *(float4*)(h + i) = o;
    v4h p = {(_Float16)o.x, (_Float16)o.y, (_Float16)o.z, (_Float16)o.w};
    *(v4h*)(hp + afrag_index(n, c, 2)) = p;
    *(float4*)(agg + i) = make_float4(0.f, 0.f, 0.f, 0.f);
}

__global__ void zero4_kernel(float* __restrict__ p, long nquads) {
    long q = (long)blockIdx.x * blockDim.x + threadIdx.x;
    if (q < nquads) *(float4*)(p + q * 4) = make_float4(0.f, 0.f, 0.f, 0.f);
}

// ---------------------------------------------------------------------------
// Global mean pool (atomic) — one block per node, 64 threads.
// ---------------------------------------------------------------------------
__global__ void pool_kernel(const float* __restrict__ h,
                            const int* __restrict__ batch,
                            float* __restrict__ sums, float* __restrict__ cnt) {
    int n = blockIdx.x;
    int g = batch[n];
    atomicAdd(&sums[(long)g * FEA + threadIdx.x], h[(long)n * FEA + threadIdx.x]);
    if (threadIdx.x == 0) atomicAdd(&cnt[g], 1.0f);
}

// ---------------------------------------------------------------------------
// MLP head: Linear(64->128) softplus, Linear(128->64) softplus, Linear(64->1)
// ---------------------------------------------------------------------------
__global__ void head_kernel(const float* __restrict__ sums,
                            const float* __restrict__ cnt,
                            const float* __restrict__ w1, const float* __restrict__ b1,
                            const float* __restrict__ w2, const float* __restrict__ b2,
                            const float* __restrict__ w3, const float* __restrict__ b3,
                            float* __restrict__ out) {
    __shared__ float pooled[64];
    __shared__ float g1[128];
    __shared__ float g2[64];
    const int g = blockIdx.x;
    const int t = threadIdx.x;
    const float c = fmaxf(cnt[g], 1.0f);
    if (t < 64) pooled[t] = sums[(long)g * FEA + t] / c;
    __syncthreads();
    float a1 = b1[t];
#pragma unroll 8
    for (int k = 0; k < 64; ++k) a1 += pooled[k] * w1[k * 128 + t];
    g1[t] = softplus_f(a1);
    __syncthreads();
    if (t < 64) {
        float a2 = b2[t];
#pragma unroll 8
        for (int k = 0; k < 128; ++k) a2 += g1[k] * w2[k * 64 + t];
        g2[t] = softplus_f(a2);
    }
    __syncthreads();
    if (t == 0) {
        float o = b3[0];
#pragma unroll 8
        for (int k = 0; k < 64; ++k) o += g2[k] * w3[k];
        out[g] = o;
    }
}

// ---------------------------------------------------------------------------
static inline char* align_up(char* p, size_t a) {
    return (char*)(((uintptr_t)p + a - 1) & ~(uintptr_t)(a - 1));
}

extern "C" void kernel_launch(void* const* d_in, const int* in_sizes, int n_in,
                              void* d_out, int out_size, void* d_ws, size_t ws_size,
                              hipStream_t stream) {
    const float* x        = (const float*)d_in[0];
    const float* eattr    = (const float*)d_in[1];
    const int*   eidx     = (const int*)d_in[2];
    const int*   batch    = (const int*)d_in[3];
    const float* emb_w    = (const float*)d_in[4];
    const float* emb_b    = (const float*)d_in[5];
    const float* conv_wf  = (const float*)d_in[6];
    const float* conv_bf  = (const float*)d_in[7];
    const float* conv_ws  = (const float*)d_in[8];
    const float* conv_bs  = (const float*)d_in[9];
    const float* bn_gamma = (const float*)d_in[10];
    const float* bn_beta  = (const float*)d_in[11];
    const float* bn_mean  = (const float*)d_in[12];
    const float* bn_var   = (const float*)d_in[13];
    const float* fc1_w    = (const float*)d_in[14];
    const float* fc1_b    = (const float*)d_in[15];
    const float* fc2_w    = (const float*)d_in[16];
    const float* fc2_b    = (const float*)d_in[17];
    const float* fc3_w    = (const float*)d_in[18];
    const float* fc3_b    = (const float*)d_in[19];
    float* out = (float*)d_out;

    const int nMt = N_NODES / 16;   // 3125 node tiles
    const int nEt = N_EDGES / 16;   // 50000 edge tiles

    // ---- workspace carve-up (256B aligned regions) ----
    char* ws = (char*)d_ws;
    float* h    = (float*)ws; ws = align_up(ws + (size_t)N_NODES * FEA * 4, 256);
    float* P    = (float*)ws; ws = align_up(ws + (size_t)N_NODES * 256 * 4, 256);
    float* agg  = (float*)ws; ws = align_up(ws + (size_t)N_NODES * FEA * 4, 256);
    float* sums = (float*)ws;                       // sums & cnt contiguous
    float* cnt  = sums + (size_t)N_GRAPHS * FEA;
    ws = align_up(ws + ((size_t)N_GRAPHS * FEA + N_GRAPHS) * 4, 256);
    _Float16* hp  = (_Float16*)ws; ws = align_up(ws + (size_t)nMt * 2 * FRAG_ELEMS * 2, 256);
    _Float16* xp  = (_Float16*)ws; ws = align_up(ws + (size_t)nMt * 3 * FRAG_ELEMS * 2, 256);
    _Float16* epk = (_Float16*)ws; ws = align_up(ws + (size_t)nEt * 2 * FRAG_ELEMS * 2, 256);
    _Float16* wembp  = (_Float16*)ws; ws = align_up(ws + (size_t)4 * 3 * FRAG_ELEMS * 2, 256);
    _Float16* wnodep = (_Float16*)ws; ws = align_up(ws + (size_t)N_CONV * 16 * 2 * FRAG_ELEMS * 2, 256);
    _Float16* wedgep = (_Float16*)ws; ws = align_up(ws + (size_t)N_CONV * 2 * 4 * 2 * FRAG_ELEMS * 2, 256);

    // ---- one-time packing (re-run every launch; deterministic) ----
    {
        long tx = (long)nMt * 3 * 32;
        pack_a_kernel<92, 3><<<(int)((tx + 255) / 256), 256, 0, stream>>>(x, xp, N_NODES);
        long te = (long)nEt * 2 * 32;
        pack_a_kernel<EDIM, 2><<<(int)((te + 255) / 256), 256, 0, stream>>>(eattr, epk, N_EDGES);
        pack_b_kernel<92, 3, 4><<<(4 * 3 * 32 + 63) / 64, 64, 0, stream>>>(emb_w, wembp, FEA, 0);
        for (int l = 0; l < N_CONV; ++l) {
            const float* wf = conv_wf + (size_t)l * ZDIM * FEA;
            const float* wsm = conv_ws + (size_t)l * ZDIM * FEA;
            _Float16* wn = wnodep + (size_t)l * 16 * 2 * FRAG_ELEMS;
            const int partSz = 4 * 2 * FRAG_ELEMS;      // 4 col tiles x 2 k-steps
            const int tb = (4 * 2 * 32 + 63) / 64;
            pack_b_kernel<64, 2, 4><<<tb, 64, 0, stream>>>(wf,  wn + 0 * partSz, FEA, 0);   // Fi
            pack_b_kernel<64, 2, 4><<<tb, 64, 0, stream>>>(wf,  wn + 1 * partSz, FEA, 64);  // Fj
            pack_b_kernel<64, 2, 4><<<tb, 64, 0, stream>>>(wsm, wn + 2 * partSz, FEA, 0);   // Si
            pack_b_kernel<64, 2, 4><<<tb, 64, 0, stream>>>(wsm, wn + 3 * partSz, FEA, 64);  // Sj
            _Float16* we = wedgep + (size_t)l * 2 * 4 * 2 * FRAG_ELEMS;
            pack_b_kernel<EDIM, 2, 4><<<tb, 64, 0, stream>>>(wf,  we, FEA, 128);
            pack_b_kernel<EDIM, 2, 4><<<tb, 64, 0, stream>>>(wsm, we + 4 * 2 * FRAG_ELEMS, FEA, 128);
        }
    }

    // ---- h = x @ emb_w + emb_b (fp32 h + f16 hpack) ----
    embed_kernel<<<nMt, 128, 0, stream>>>(xp, wembp, emb_b, h, hp);

    // ---- zero agg / sums / cnt ----
    {
        long qa = (long)N_NODES * FEA / 4;
        zero4_kernel<<<(int)((qa + 255) / 256), 256, 0, stream>>>(agg, qa);
        long qs = ((long)N_GRAPHS * FEA + N_GRAPHS) / 4;   // 16640 floats, divisible by 4
        zero4_kernel<<<(int)((qs + 255) / 256), 256, 0, stream>>>(sums, qs);
    }

    // ---- conv layers ----
    for (int l = 0; l < N_CONV; ++l) {
        const _Float16* wn = wnodep + (size_t)l * 16 * 2 * FRAG_ELEMS;
        const _Float16* we = wedgep + (size_t)l * 2 * 4 * 2 * FRAG_ELEMS;
        const float* bf = conv_bf + (size_t)l * FEA;
        const float* bs = conv_bs + (size_t)l * FEA;

        // 3125*16 waves, 8 waves/block -> exact grid
        node_gemm_kernel<<<nMt * 16 / 8, 256, 0, stream>>>(hp, wn, bf, bs, P);

        // 50000 waves (one per 16-edge tile), 8 waves/block -> exact grid
        edge_kernel<<<nEt / 8, 256, 0, stream>>>(epk, eidx, we, we + 4 * 2 * FRAG_ELEMS, P, agg);

        bn_residual_kernel<<<N_NODES * 16 / 256, 256, 0, stream>>>(
            h, hp, agg,
            bn_gamma + (size_t)l * FEA, bn_beta + (size_t)l * FEA,
            bn_mean + (size_t)l * FEA, bn_var + (size_t)l * FEA);
    }

    // ---- global mean pool + MLP head ----
    pool_kernel<<<N_NODES, 64, 0, stream>>>(h, batch, sums, cnt);
    head_kernel<<<N_GRAPHS, 128, 0, stream>>>(sums, cnt, fc1_w, fc1_b,
                                              fc2_w, fc2_b, fc3_w, fc3_b, out);
}
